// SwitchableAttention_34213709480680
// MI455X (gfx1250) — compile-verified, matrix operand order checked by reading, not run on hardware
//
#include <hip/hip_runtime.h>
#include <hip/hip_bf16.h>

typedef _Float16 f16;
typedef __attribute__((ext_vector_type(16))) _Float16 v16h;
typedef __attribute__((ext_vector_type(8)))  _Float16 v8h;
typedef __attribute__((ext_vector_type(8)))  float    v8f;
typedef __attribute__((ext_vector_type(4)))  float    v4f;
typedef __attribute__((ext_vector_type(4)))  unsigned u32x4;
typedef __attribute__((ext_vector_type(8)))  unsigned u32x8;

#define DIN    512
#define HEADS  8
#define DHEAD  64
#define DIM    512
#define NMAX   512
#define NBATCH 32
#define LN_EPS 1e-5f

// ---------------------------------------------------------------------------
// WMMA helpers (wave32, 16x16x32 f16 -> f32)
// ---------------------------------------------------------------------------
__device__ __forceinline__ v8f wmma_f16(v16h a, v16h b, v8f c) {
  return __builtin_amdgcn_wmma_f32_16x16x32_f16(false, a, false, b,
                                                (short)0, c, false, false);
}

// Load a 16x32 A-fragment (or B-fragment given a transposed/col-major view)
// from row-major f16 memory with leading dimension ld (halves).
// Lane mapping per ISA 7.12.2: row = lane%16, K = 8*(lane/16)+j (j<8) and
// K = 16 + 8*(lane/16)+(j-8) (j>=8) -> two contiguous 16B chunks per lane.
__device__ __forceinline__ v16h frag_ld(const f16* p, int ld) {
  const int lane = threadIdx.x & 31;
  const int row  = lane & 15;
  const int half = lane >> 4;
  const f16* q = p + row * ld + half * 8;
  v8h lo = *(const v8h*)(q);
  v8h hi = *(const v8h*)(q + 16);
  v16h r;
#pragma unroll
  for (int j = 0; j < 8; ++j) { r[j] = lo[j]; r[j + 8] = hi[j]; }
  return r;
}

__device__ __forceinline__ v8f v8f_zero() {
  v8f z = {0.f, 0.f, 0.f, 0.f, 0.f, 0.f, 0.f, 0.f};
  return z;
}

__device__ __forceinline__ int lower_bound_i32(const int* a, int n, int v) {
  int lo = 0, hi = n;
  while (lo < hi) { int mid = (lo + hi) >> 1; if (a[mid] < v) lo = mid + 1; else hi = mid; }
  return lo;
}

// ---------------------------------------------------------------------------
// CDNA5 async data movers
// ---------------------------------------------------------------------------
// Per-lane async global->LDS copy of 16 bytes (ASYNCcnt-tracked).
__device__ __forceinline__ void async_g2l_b128(f16* lds_dst, const f16* gsrc) {
  unsigned ldsoff = (unsigned)(unsigned long long)lds_dst;  // as(3) offset
  asm volatile("global_load_async_to_lds_b128 %0, %1, off"
               :
               : "v"(ldsoff), "v"(gsrc)
               : "memory");
}
__device__ __forceinline__ void wait_async0() {
  asm volatile("s_wait_asynccnt 0x0" ::: "memory");
}

// TDM: DMA a 2D tile (tile_d0 elems/row x tile_d1 rows of 2-byte data,
// row stride `stride0` elements) from global memory to LDS (contiguous).
// Wave-level op (EXEC ignored) -> call from one wave only; TENSORcnt-tracked.
__device__ __forceinline__ void tdm_load_tile_2d_f16(unsigned lds_off,
                                                     const f16* gaddr,
                                                     unsigned tile_d0,
                                                     unsigned tile_d1,
                                                     unsigned tensor_d0,
                                                     unsigned tensor_d1,
                                                     unsigned stride0) {
  const unsigned long long ga = (unsigned long long)gaddr;
  u32x4 g0;
  g0[0] = 1u;                                        // count=1, user mode
  g0[1] = lds_off;                                   // lds_addr
  g0[2] = (unsigned)(ga & 0xffffffffu);              // global_addr[31:0]
  g0[3] = (unsigned)((ga >> 32) & 0x01ffffffu)       // global_addr[56:32]
          | (2u << 30);                              // type=2 ("image")
  u32x8 g1;
  g1[0] = (1u << 16);                                // data_size=1 (2 bytes)
  g1[1] = (tensor_d0 & 0xffffu) << 16;               // tensor_dim0[15:0]
  g1[2] = (tensor_d0 >> 16) | ((tensor_d1 & 0xffffu) << 16);
  g1[3] = (tensor_d1 >> 16) | (tile_d0 << 16);       // tile_dim0
  g1[4] = tile_d1;                                   // tile_dim1 (tile_dim2=0)
  g1[5] = stride0;                                   // tensor_dim0_stride[31:0]
  g1[6] = 0u;                                        // stride0[47:32], stride1 lo
  g1[7] = 0u;                                        // tensor_dim1_stride hi
  asm volatile("tensor_load_to_lds %0, %1"
               :
               : "s"(g0), "s"(g1)
               : "memory");
}

// ---------------------------------------------------------------------------
// Kernel 1: per-graph counts (src is sorted)
// ---------------------------------------------------------------------------
__global__ void k_counts(const int* __restrict__ src, int ntot,
                         int* __restrict__ counts) {
  int b = threadIdx.x;
  if (b < NBATCH) {
    int lb = lower_bound_i32(src, ntot, b);
    int ub = lower_bound_i32(src, ntot, b + 1);
    counts[b] = ub - lb;
  }
}

// ---------------------------------------------------------------------------
// Kernel 2: LayerNorm + scatter to dense f16 [B*NMAX, DIN]; one wave per node
// ---------------------------------------------------------------------------
__global__ void k_ln_scatter(const float* __restrict__ z,
                             const int* __restrict__ src,
                             const float* __restrict__ gamma,
                             const float* __restrict__ beta,
                             int ntot,
                             f16* __restrict__ xdense,
                             int* __restrict__ rowmap) {
  const int wave = threadIdx.x >> 5;
  const int lane = threadIdx.x & 31;
  const int i = blockIdx.x * 8 + wave;
  if (i >= ntot) return;

  const int b = src[i];
  const int pos = i - lower_bound_i32(src, ntot, b);
  const int drow = b * NMAX + pos;
  if (lane == 0) rowmap[i] = drow;

  const float* zi = z + (long)i * DIN;
  float vals[16];
  float s = 0.f;
#pragma unroll
  for (int j = 0; j < 4; ++j) {
    v4f t = *(const v4f*)(zi + lane * 16 + j * 4);
#pragma unroll
    for (int e = 0; e < 4; ++e) { vals[4 * j + e] = t[e]; s += t[e]; }
  }
#pragma unroll
  for (int off = 16; off >= 1; off >>= 1) s += __shfl_xor(s, off, 32);
  const float mu = s * (1.0f / DIN);

  float ss = 0.f;
#pragma unroll
  for (int j = 0; j < 16; ++j) { float d = vals[j] - mu; ss += d * d; }
#pragma unroll
  for (int off = 16; off >= 1; off >>= 1) ss += __shfl_xor(ss, off, 32);
  const float rstd = rsqrtf(ss * (1.0f / DIN) + LN_EPS);

  f16* dst = xdense + (long)drow * DIN + lane * 16;
  v8h o0, o1;
#pragma unroll
  for (int j = 0; j < 16; ++j) {
    const int c = lane * 16 + j;
    float nv = (vals[j] - mu) * rstd * gamma[c] + beta[c];
    if (j < 8) o0[j] = (f16)nv; else o1[j - 8] = (f16)nv;
  }
  *(v8h*)(dst) = o0;
  *(v8h*)(dst + 8) = o1;
}

// ---------------------------------------------------------------------------
// Kernel 3: Q/K/V projections. x[16384,512] @ W[512,512] -> f16.
// grid: (rows/64, DIM/64, 3); block 128 (4 waves); z selects Wq/Wk/Wv.
// A tile staged by the Tensor Data Mover; W tile transposed+converted by VALU.
// q,k stored [B,H,N,D]; v stored transposed [B,H,D,N].
// ---------------------------------------------------------------------------
__global__ void k_qkv(const f16* __restrict__ xdense,
                      const float* __restrict__ Wq,
                      const float* __restrict__ Wk,
                      const float* __restrict__ Wv,
                      f16* __restrict__ qo, f16* __restrict__ ko,
                      f16* __restrict__ vto) {
  __shared__ __align__(16) f16 As[64 * 32];  // A tile, row-major
  __shared__ __align__(16) f16 Bs[64 * 32];  // W tile transposed: [n][k]
  const int row0 = blockIdx.x * 64;
  const int col0 = blockIdx.y * 64;
  const int which = blockIdx.z;
  const float* W = (which == 0) ? Wq : (which == 1) ? Wk : Wv;
  const int t = threadIdx.x;
  const int wave = t >> 5, lane = t & 31;
  const int lanecol = lane & 15, half = lane >> 4;
  const unsigned as_off = (unsigned)(unsigned long long)(&As[0]);

  v8f acc[4] = {v8f_zero(), v8f_zero(), v8f_zero(), v8f_zero()};

  for (int k0 = 0; k0 < DIN; k0 += 32) {
    if (wave == 0) {  // TDM: stage 64x32 f16 A tile, wave-level DMA
      tdm_load_tile_2d_f16(as_off, xdense + (long)row0 * DIN + k0,
                           /*tile_d0=*/32, /*tile_d1=*/64,
                           /*tensor_d0=*/DIN, /*tensor_d1=*/NBATCH * NMAX,
                           /*stride0=*/DIN);
      __builtin_amdgcn_s_wait_tensorcnt(0);
    }
    {  // stage W transposed + f32->f16
      const int kl = t >> 2;            // 0..31
      const int nb = (t & 3) * 16;      // 0,16,32,48
      const float* wp = W + (long)(k0 + kl) * DIM + col0 + nb;
#pragma unroll
      for (int j = 0; j < 16; ++j) Bs[(nb + j) * 32 + kl] = (f16)wp[j];
    }
    __syncthreads();
    v16h a = frag_ld(As + (wave * 16) * 32, 32);
#pragma unroll
    for (int c = 0; c < 4; ++c) {
      v16h bf = frag_ld(Bs + (c * 16) * 32, 32);
      acc[c] = wmma_f16(a, bf, acc[c]);
    }
    __syncthreads();
  }

#pragma unroll
  for (int c = 0; c < 4; ++c) {
    const int col = col0 + c * 16 + lanecol;
    const int hh = col >> 6, d = col & 63;
#pragma unroll
    for (int r = 0; r < 8; ++r) {
      const int grow = row0 + wave * 16 + r + 8 * half;
      const int bb = grow >> 9, n = grow & (NMAX - 1);
      const f16 val = (f16)acc[c][r];
      const long bh = (long)(bb * HEADS + hh);
      if (which == 0)      qo[(bh * NMAX + n) * DHEAD + d] = val;
      else if (which == 1) ko[(bh * NMAX + n) * DHEAD + d] = val;
      else                 vto[(bh * DHEAD + d) * NMAX + n] = val;
    }
  }
}

// ---------------------------------------------------------------------------
// Kernel 4: flash attention. grid (NMAX/128, H, B), block 256 (8 waves).
// Each wave: 16 query rows, streams 32-key chunks, P through per-wave LDS.
// ---------------------------------------------------------------------------
__global__ void k_attn(const f16* __restrict__ q, const f16* __restrict__ k,
                       const f16* __restrict__ vt,
                       const int* __restrict__ counts,
                       f16* __restrict__ om) {
  __shared__ __align__(16) f16 P[8][16 * 32];  // per-wave P tile [16 q][32 keys]
  const int wave = threadIdx.x >> 5, lane = threadIdx.x & 31;
  const int lanecol = lane & 15, half = lane >> 4;
  const int b = blockIdx.z, h = blockIdx.y;
  const int q0 = blockIdx.x * 128 + wave * 16;
  const long bh = (long)(b * HEADS + h);
  const f16* Q = q  + bh * NMAX * DHEAD;
  const f16* K = k  + bh * NMAX * DHEAD;
  const f16* V = vt + bh * DHEAD * NMAX;

  const v16h aq0 = frag_ld(Q + q0 * DHEAD,      DHEAD);  // d = 0..31
  const v16h aq1 = frag_ld(Q + q0 * DHEAD + 32, DHEAD);  // d = 32..63

  float mrun[8], lrun[8], alpha[8];
#pragma unroll
  for (int r = 0; r < 8; ++r) { mrun[r] = -__builtin_inff(); lrun[r] = 0.f; }
  v8f o[4] = {v8f_zero(), v8f_zero(), v8f_zero(), v8f_zero()};

  const int cnt = counts[b];
  const int nch = (cnt + 31) >> 5;
  const float sc = 0.125f;  // DHEAD^-0.5

  for (int kt = 0; kt < nch; ++kt) {
    const int key0 = kt * 32;
    v8f s0 = v8f_zero(), s1 = v8f_zero();
    {
      v16h bk;
      bk = frag_ld(K + (key0)      * DHEAD,      DHEAD); s0 = wmma_f16(aq0, bk, s0);
      bk = frag_ld(K + (key0)      * DHEAD + 32, DHEAD); s0 = wmma_f16(aq1, bk, s0);
      bk = frag_ld(K + (key0 + 16) * DHEAD,      DHEAD); s1 = wmma_f16(aq0, bk, s1);
      bk = frag_ld(K + (key0 + 16) * DHEAD + 32, DHEAD); s1 = wmma_f16(aq1, bk, s1);
    }
    const int keyA = key0 + lanecol;
    const int keyB = key0 + 16 + lanecol;
    float cmax[8];
#pragma unroll
    for (int r = 0; r < 8; ++r) {
      float a0 = (keyA < cnt) ? s0[r] * sc : -__builtin_inff();
      float a1 = (keyB < cnt) ? s1[r] * sc : -__builtin_inff();
      s0[r] = a0; s1[r] = a1;
      cmax[r] = fmaxf(a0, a1);
    }
#pragma unroll
    for (int off = 8; off >= 1; off >>= 1)
#pragma unroll
      for (int r = 0; r < 8; ++r)
        cmax[r] = fmaxf(cmax[r], __shfl_xor(cmax[r], off, 32));

    float rs[8];
#pragma unroll
    for (int r = 0; r < 8; ++r) {
      const float mn = fmaxf(mrun[r], cmax[r]);
      alpha[r] = __expf(mrun[r] - mn);      // mrun=-inf -> 0
      const float p0 = __expf(s0[r] - mn);  // masked -inf -> 0
      const float p1 = __expf(s1[r] - mn);
      s0[r] = p0; s1[r] = p1;
      rs[r] = p0 + p1;
      mrun[r] = mn;
    }
#pragma unroll
    for (int off = 8; off >= 1; off >>= 1)
#pragma unroll
      for (int r = 0; r < 8; ++r) rs[r] += __shfl_xor(rs[r], off, 32);
#pragma unroll
    for (int r = 0; r < 8; ++r) lrun[r] = lrun[r] * alpha[r] + rs[r];

    // stage P (16x32) into this wave's LDS tile, re-load as A fragment
    f16* Pw = &P[wave][0];
#pragma unroll
    for (int r = 0; r < 8; ++r) {
      const int m = r + 8 * half;
      Pw[m * 32 + lanecol]      = (f16)s0[r];
      Pw[m * 32 + 16 + lanecol] = (f16)s1[r];
    }
    asm volatile("s_wait_dscnt 0x0" ::: "memory");
    const v16h ap = frag_ld(Pw, 32);
#pragma unroll
    for (int c = 0; c < 4; ++c) {
#pragma unroll
      for (int r = 0; r < 8; ++r) o[c][r] *= alpha[r];
      v16h bv = frag_ld(V + (c * 16) * NMAX + key0, NMAX);  // Vt[d][key]
      o[c] = wmma_f16(ap, bv, o[c]);
    }
  }

#pragma unroll
  for (int r = 0; r < 8; ++r) {
    const float inv = (lrun[r] > 0.f) ? 1.0f / lrun[r] : 0.f;
#pragma unroll
    for (int c = 0; c < 4; ++c) o[c][r] *= inv;
  }
#pragma unroll
  for (int c = 0; c < 4; ++c) {
    const int col = h * DHEAD + c * 16 + lanecol;
#pragma unroll
    for (int r = 0; r < 8; ++r) {
      const int m = q0 + r + 8 * half;
      om[((long)(b * NMAX + m)) * DIM + col] = (f16)o[c][r];
    }
  }
}

// ---------------------------------------------------------------------------
// Kernel 5: gather + output projection + bias, f32 output [ntot, DIM].
// Gathered A rows staged with per-lane async global->LDS b128 copies.
// ---------------------------------------------------------------------------
__global__ void k_outproj(const f16* __restrict__ om,
                          const int* __restrict__ rowmap,
                          const float* __restrict__ Wo,
                          const float* __restrict__ bo,
                          int ntot, float* __restrict__ out) {
  __shared__ __align__(16) f16 As[64 * 32];
  __shared__ __align__(16) f16 Bs[64 * 32];
  const int row0 = blockIdx.x * 64;
  const int col0 = blockIdx.y * 64;
  const int t = threadIdx.x;
  const int wave = t >> 5, lane = t & 31;
  const int lanecol = lane & 15, half = lane >> 4;

  v8f acc[4] = {v8f_zero(), v8f_zero(), v8f_zero(), v8f_zero()};

  for (int k0 = 0; k0 < DIM; k0 += 32) {
    {  // stage gathered A rows via async global->LDS copies
      const int r = t >> 1, cc = (t & 1) * 16;
      const int grow = row0 + r;
      if (grow < ntot) {
        const f16* sp = om + (long)rowmap[grow] * DIM + k0 + cc;
        async_g2l_b128(As + r * 32 + cc,     sp);
        async_g2l_b128(As + r * 32 + cc + 8, sp + 8);
      } else {
#pragma unroll
        for (int j = 0; j < 16; ++j) As[r * 32 + cc + j] = (f16)0.f;
      }
    }
    {  // stage Wo transposed + f32->f16
      const int kl = t >> 2;
      const int nb = (t & 3) * 16;
      const float* wp = Wo + (long)(k0 + kl) * DIM + col0 + nb;
#pragma unroll
      for (int j = 0; j < 16; ++j) Bs[(nb + j) * 32 + kl] = (f16)wp[j];
    }
    wait_async0();
    __syncthreads();
    v16h a = frag_ld(As + (wave * 16) * 32, 32);
#pragma unroll
    for (int c = 0; c < 4; ++c) {
      v16h bf = frag_ld(Bs + (c * 16) * 32, 32);
      acc[c] = wmma_f16(a, bf, acc[c]);
    }
    __syncthreads();
  }

#pragma unroll
  for (int c = 0; c < 4; ++c) {
    const int col = col0 + c * 16 + lanecol;
    const float bias = bo[col];
#pragma unroll
    for (int r = 0; r < 8; ++r) {
      const int grow = row0 + wave * 16 + r + 8 * half;
      if (grow < ntot) out[(long)grow * DIM + col] = acc[c][r] + bias;
    }
  }
}

// ---------------------------------------------------------------------------
extern "C" void kernel_launch(void* const* d_in, const int* in_sizes, int n_in,
                              void* d_out, int out_size, void* d_ws, size_t ws_size,
                              hipStream_t stream) {
  const float* z     = (const float*)d_in[0];
  const int*   src   = (const int*)d_in[1];
  const float* gamma = (const float*)d_in[2];
  const float* beta  = (const float*)d_in[3];
  const float* Wq    = (const float*)d_in[4];
  const float* Wk    = (const float*)d_in[5];
  const float* Wv    = (const float*)d_in[6];
  const float* Wo    = (const float*)d_in[7];
  const float* bo    = (const float*)d_in[8];
  const int ntot = in_sizes[1];
  float* out = (float*)d_out;

  // workspace carve (256B-aligned regions)
  char* ws = (char*)d_ws;
  size_t off = 0;
  auto carve = [&](size_t bytes) -> void* {
    void* p = ws + off;
    off += (bytes + 255) & ~(size_t)255;
    return p;
  };
  const size_t dense_halves = (size_t)NBATCH * NMAX * DIN;           // 8M halves
  f16* xdense = (f16*)carve(dense_halves * sizeof(f16));
  f16* qb     = (f16*)carve(dense_halves * sizeof(f16));             // [B,H,N,D]
  f16* kb     = (f16*)carve(dense_halves * sizeof(f16));             // [B,H,N,D]
  f16* vtb    = (f16*)carve(dense_halves * sizeof(f16));             // [B,H,D,N]
  f16* om     = (f16*)carve(dense_halves * sizeof(f16));             // [B*N, DIM]
  int* counts = (int*)carve(NBATCH * sizeof(int));
  int* rowmap = (int*)carve((size_t)ntot * sizeof(int));

  hipMemsetAsync(xdense, 0, dense_halves * sizeof(f16), stream);

  k_counts<<<1, 32, 0, stream>>>(src, ntot, counts);
  k_ln_scatter<<<(ntot + 7) / 8, 256, 0, stream>>>(z, src, gamma, beta, ntot,
                                                   xdense, rowmap);
  k_qkv<<<dim3((NBATCH * NMAX) / 64, DIM / 64, 3), 128, 0, stream>>>(
      xdense, Wq, Wk, Wv, qb, kb, vtb);
  k_attn<<<dim3(NMAX / 128, HEADS, NBATCH), 256, 0, stream>>>(qb, kb, vtb,
                                                              counts, om);
  k_outproj<<<dim3((ntot + 63) / 64, DIM / 64), 128, 0, stream>>>(
      om, rowmap, Wo, bo, ntot, out);
}
